// KNN_30081950941621
// MI455X (gfx1250) — matrix operand order repaired and checked by reading, not compiled, not run on hardware
//
#include <hip/hip_runtime.h>
#include <hip/hip_bf16.h>

typedef __attribute__((ext_vector_type(16))) _Float16 v16h;
typedef __attribute__((ext_vector_type(8)))  _Float16 v8h;
typedef __attribute__((ext_vector_type(8)))  float    v8f;

#define KNN_B 512
#define KNN_D 64
#define KNN_N 200000
#define KNN_K 16
#define KNN_L 10

#define NCHUNK 125
#define TILES_PER_CHUNK 100   // 125 * 100 * 16 = 200000
#define QPAIRS 16             // 512 / 32 : each block serves 32 queries
#define NWAVES 4
#define NCAND (NCHUNK * 16)   // 2000 candidates per query at merge
#define WS_ENTRIES (KNN_B * NCAND)  // 1,024,000 entries (dist + idx arrays)

__device__ __forceinline__ float knn_inf() { return __builtin_inff(); }

// ---------------------------------------------------------------------------
// Kernel 0: one-time f32 -> f16 conversion of the train matrix (row-major,
// same layout). 25.6MB result stays L2-resident for the GEMM kernel, halving
// its bandwidth need and removing all per-tile cvt work.
// ---------------------------------------------------------------------------
__global__ __launch_bounds__(256) void knn_cvt_kernel(const float* __restrict__ T,
                                                      _Float16* __restrict__ Tf16) {
  size_t i = (size_t)blockIdx.x * blockDim.x + threadIdx.x;   // 8 elements each
  if (i >= (size_t)KNN_N * KNN_D / 8) return;
  const float4* p = (const float4*)(T + i * 8);
  float4 a = p[0], b = p[1];
  v8h h;
  h[0] = (_Float16)a.x; h[1] = (_Float16)a.y; h[2] = (_Float16)a.z; h[3] = (_Float16)a.w;
  h[4] = (_Float16)b.x; h[5] = (_Float16)b.y; h[6] = (_Float16)b.z; h[7] = (_Float16)b.w;
  *(v8h*)(Tf16 + i * 8) = h;
}

// ---------------------------------------------------------------------------
// Kernel 1: train row norms t2[n] = sum_d T[n][d]^2 (full f32 precision).
// ---------------------------------------------------------------------------
__global__ __launch_bounds__(256) void knn_t2_kernel(const float* __restrict__ T,
                                                     float* __restrict__ t2) {
  int n = blockIdx.x * blockDim.x + threadIdx.x;
  if (n >= KNN_N) return;
  const float4* r = (const float4*)(T + (size_t)n * KNN_D);
  float s = 0.f;
#pragma unroll
  for (int i = 0; i < KNN_D / 4; ++i) {
    float4 v = r[i];
    s = fmaf(v.x, v.x, s);
    s = fmaf(v.y, v.y, s);
    s = fmaf(v.z, v.z, s);
    s = fmaf(v.w, v.w, s);
  }
  t2[n] = s;
}

// Insert candidate (s, idx) into a 16-deep "keep the smallest" list kept in
// LDS; cmax/cpos are the owner lane's cached current max entry.
__device__ __forceinline__ void topk_insert(float* dv, int* iv,
                                            float& cmax, int& cpos,
                                            float s, int idx) {
  if (s < cmax) {
    dv[cpos] = s;
    iv[cpos] = idx;
    float m = dv[0];
    int p = 0;
#pragma unroll
    for (int j = 1; j < KNN_K; ++j) {
      float x = dv[j];
      if (x > m) { m = x; p = j; }
    }
    cmax = m;
    cpos = p;
  }
}

// ---------------------------------------------------------------------------
// Kernel 2: fused WMMA distance GEMM + per-chunk top-16.
// grid = (NCHUNK, QPAIRS); block = 4 waves (128 threads).
// Each wave owns TWO 16-query tiles (32 queries): one f16 B fragment (2x 32B
// loads) feeds 4 v_wmma_f32_16x16x32_f16; during selection all 32 lanes own
// a query (lanes 0-15 -> tile 0, lanes 16-31 -> tile 1).
// ---------------------------------------------------------------------------
__global__ __launch_bounds__(128) void knn_main_kernel(const float* __restrict__ X,
                                                       const _Float16* __restrict__ Tf16,
                                                       const float* __restrict__ t2,
                                                       float* __restrict__ wsD,
                                                       int* __restrict__ wsI) {
  __shared__ float stag [NWAVES][2][16][16];   // [wave][qtile][query][traincol]
  __shared__ float listD[NWAVES][32][KNN_K];
  __shared__ int   listI[NWAVES][32][KNN_K];

  const int lane = threadIdx.x & 31;
  const int wave = threadIdx.x >> 5;
  const int m    = lane & 15;   // row-in-tile (A) / col-in-tile (B)
  const int hf   = lane >> 4;   // half-wave select
  const int chunk = blockIdx.x;
  const int qbase = blockIdx.y * 32;

  // ---- A fragments for both query tiles (contiguous K order: k = 16*hf + s, matching B).
  const float* xrow0 = X + (size_t)(qbase + m) * KNN_D;
  const float* xrow1 = X + (size_t)(qbase + 16 + m) * KNN_D;
  v16h a0, a1, a2, a3;
#pragma unroll
  for (int v = 0; v < 8; ++v) {
    int kb = (hf << 4) + (v << 1);
    a0[2 * v]     = (_Float16)xrow0[kb];
    a0[2 * v + 1] = (_Float16)xrow0[kb + 1];
    a1[2 * v]     = (_Float16)xrow0[kb + 32];
    a1[2 * v + 1] = (_Float16)xrow0[kb + 33];
    a2[2 * v]     = (_Float16)xrow1[kb];
    a2[2 * v + 1] = (_Float16)xrow1[kb + 1];
    a3[2 * v]     = (_Float16)xrow1[kb + 32];
    a3[2 * v + 1] = (_Float16)xrow1[kb + 33];
  }

  // ---- init per-wave per-query top-16 lists (every lane owns one query)
#pragma unroll
  for (int j = 0; j < KNN_K; ++j) {
    listD[wave][lane][j] = knn_inf();
    listI[wave][lane][j] = 0;
  }
  float cmax = knn_inf();
  int   cpos = 0;
  float* const myD = &listD[wave][lane][0];
  int*   const myI = &listI[wave][lane][0];
  const int tsel = hf;            // which staged qtile this lane scans
  const int qloc = m;             // which query row it owns

  const int tile0 = chunk * TILES_PER_CHUNK;
  const int tileE = tile0 + TILES_PER_CHUNK;

  for (int t = tile0 + wave; t < tileE; t += NWAVES) {
    const int nbase = t * 16;
    const int n     = nbase + m;

    // B fragments (f16, pre-converted): lanes 0-15 hold K=0..15 (+32),
    // lanes 16-31 hold K=16..31 (+32). Two contiguous 32B loads per lane.
    const _Float16* trow = Tf16 + (size_t)n * KNN_D + hf * 16;
    __builtin_prefetch(trow + NWAVES * 16 * KNN_D, 0, 1);   // next tile for this wave
    v16h b0 = *(const v16h*)(trow);
    v16h b1 = *(const v16h*)(trow + 32);

    // ---- 4 WMMAs: two query tiles x two K-halves, f32 accumulate.
    v8f acc0 = {}, acc1 = {};
    acc0 = __builtin_amdgcn_wmma_f32_16x16x32_f16(false, a0, false, b0,
                                                  (short)0, acc0, false, false);
    acc1 = __builtin_amdgcn_wmma_f32_16x16x32_f16(false, a2, false, b0,
                                                  (short)0, acc1, false, false);
    acc0 = __builtin_amdgcn_wmma_f32_16x16x32_f16(false, a1, false, b1,
                                                  (short)0, acc0, false, false);
    acc1 = __builtin_amdgcn_wmma_f32_16x16x32_f16(false, a3, false, b1,
                                                  (short)0, acc1, false, false);

    // s = t2[n] - 2*dot  (||x||^2 is a per-query constant; added later)
    const float t2v = t2[n];
#pragma unroll
    for (int r = 0; r < 8; ++r) {
      stag[wave][0][r + 8 * hf][m] = fmaf(-2.f, acc0[r], t2v);
      stag[wave][1][r + 8 * hf][m] = fmaf(-2.f, acc1[r], t2v);
    }
    asm volatile("s_wait_dscnt 0" ::: "memory");  // wave-local LDS RAW

    // ---- selection: every lane owns one query row of one staged tile.
    const float4* sp = (const float4*)&stag[wave][tsel][qloc][0];
    float4 s0 = sp[0], s1 = sp[1], s2 = sp[2], s3 = sp[3];
    float mn = fminf(fminf(fminf(s0.x, s0.y), fminf(s0.z, s0.w)),
                     fminf(fminf(s1.x, s1.y), fminf(s1.z, s1.w)));
    mn = fminf(mn, fminf(fminf(fminf(s2.x, s2.y), fminf(s2.z, s2.w)),
                         fminf(fminf(s3.x, s3.y), fminf(s3.z, s3.w))));
    if (mn < cmax) {
      float sv[16] = {s0.x, s0.y, s0.z, s0.w, s1.x, s1.y, s1.z, s1.w,
                      s2.x, s2.y, s2.z, s2.w, s3.x, s3.y, s3.z, s3.w};
#pragma unroll
      for (int c = 0; c < 16; ++c) {
        topk_insert(myD, myI, cmax, cpos, sv[c], nbase + c);
      }
    }
  }

  __syncthreads();

  // ---- merge the 4 per-wave lists for each query; threads 0..31 own queries.
  const int tid = threadIdx.x;
  if (tid < 32) {
    float* dv = &listD[0][tid][0];
    int*   iv = &listI[0][tid][0];
    float mx = dv[0];
    int   mp = 0;
#pragma unroll
    for (int j = 1; j < KNN_K; ++j) {
      float x = dv[j];
      if (x > mx) { mx = x; mp = j; }
    }
#pragma unroll
    for (int w = 1; w < NWAVES; ++w) {
      for (int j = 0; j < KNN_K; ++j) {
        topk_insert(dv, iv, mx, mp, listD[w][tid][j], listI[w][tid][j]);
      }
    }
    const int Q = qbase + tid;
    size_t base = (size_t)Q * NCAND + chunk * 16;
#pragma unroll
    for (int j = 0; j < KNN_K; ++j) {
      wsD[base + j] = dv[j];
      wsI[base + j] = iv[j];
    }
  }
}

// ---------------------------------------------------------------------------
// Kernel 3: per-query merge of 125 chunk lists (2000 candidates) -> global
// top-16, then inverse-distance weighted vote + argmax. One WG per query.
// ---------------------------------------------------------------------------
__global__ __launch_bounds__(256) void knn_final_kernel(const float* __restrict__ X,
                                                        const int* __restrict__ labels,
                                                        const float* __restrict__ wsD,
                                                        const int* __restrict__ wsI,
                                                        float* __restrict__ out) {
  __shared__ float cd[NCAND];
  __shared__ int   ci[NCAND];
  __shared__ float redD[256];
  __shared__ int   redI[256];
  __shared__ float selD[KNN_K];
  __shared__ int   selI[KNN_K];
  __shared__ float x2s;

  const int Q   = blockIdx.x;
  const int tid = threadIdx.x;

  // ||x||^2
  float p = 0.f;
  if (tid < KNN_D) {
    float v = X[(size_t)Q * KNN_D + tid];
    p = v * v;
  }
  redD[tid] = p;
  __syncthreads();
  for (int s = 128; s > 0; s >>= 1) {
    if (tid < s) redD[tid] += redD[tid + s];
    __syncthreads();
  }
  if (tid == 0) x2s = redD[0];
  __syncthreads();

  // load candidates for this query (fully coalesced: flat per-query layout)
  for (int c = tid; c < NCAND; c += 256) {
    size_t off = (size_t)Q * NCAND + c;
    cd[c] = wsD[off];
    ci[c] = wsI[off];
  }
  __syncthreads();

  // 16 x (parallel argmin over 2000, then knock out the winner)
  for (int it = 0; it < KNN_K; ++it) {
    float bm = knn_inf();
    int   bi = -1;
    for (int c = tid; c < NCAND; c += 256) {
      float v = cd[c];
      if (v < bm) { bm = v; bi = c; }
    }
    redD[tid] = bm;
    redI[tid] = bi;
    __syncthreads();
    for (int s = 128; s > 0; s >>= 1) {
      if (tid < s) {
        if (redD[tid + s] < redD[tid]) {
          redD[tid] = redD[tid + s];
          redI[tid] = redI[tid + s];
        }
      }
      __syncthreads();
    }
    if (tid == 0) {
      int w = redI[0];
      selD[it] = redD[0];
      selI[it] = (w >= 0) ? ci[w] : 0;
      if (w >= 0) cd[w] = knn_inf();
    }
    __syncthreads();
  }

  if (tid == 0) {
    const float x2 = x2s;
    float dist[KNN_K];
    bool anyZero = false;
#pragma unroll
    for (int j = 0; j < KNN_K; ++j) {
      float d2 = selD[j] + x2;
      d2 = fmaxf(d2, 0.f);
      dist[j] = sqrtf(d2);
      if (dist[j] == 0.f) anyZero = true;
    }
    float votes[KNN_L];
#pragma unroll
    for (int l = 0; l < KNN_L; ++l) votes[l] = 0.f;
#pragma unroll
    for (int j = 0; j < KNN_K; ++j) {
      float w = anyZero ? (dist[j] == 0.f ? 1.f : 0.f) : (1.f / dist[j]);
      int lab = labels[selI[j]];
      votes[lab] += w;
    }
    int best = 0;
    float bv = votes[0];
#pragma unroll
    for (int l = 1; l < KNN_L; ++l) {
      if (votes[l] > bv) { bv = votes[l]; best = l; }
    }
    out[Q] = (float)best;
  }
}

// ---------------------------------------------------------------------------
extern "C" void kernel_launch(void* const* d_in, const int* in_sizes, int n_in,
                              void* d_out, int out_size, void* d_ws, size_t ws_size,
                              hipStream_t stream) {
  (void)in_sizes; (void)n_in; (void)out_size; (void)ws_size;
  const float* X      = (const float*)d_in[0];
  const float* T      = (const float*)d_in[1];
  const int*   labels = (const int*)d_in[2];
  // d_in[3] = k (16), d_in[4] = num_labels (10): compile-time constants here.

  float*    wsD  = (float*)d_ws;                  // [B][NCAND]
  int*      wsI  = (int*)(wsD + WS_ENTRIES);      // same shape
  float*    t2   = (float*)(wsI + WS_ENTRIES);    // [N]
  _Float16* Tf16 = (_Float16*)(t2 + KNN_N);       // [N][D] f16 copy
  float*    out  = (float*)d_out;                 // [512]

  knn_cvt_kernel<<<((KNN_N * KNN_D / 8) + 255) / 256, 256, 0, stream>>>(T, Tf16);
  knn_t2_kernel<<<(KNN_N + 255) / 256, 256, 0, stream>>>(T, t2);
  knn_main_kernel<<<dim3(NCHUNK, QPAIRS), NWAVES * 32, 0, stream>>>(X, Tf16, t2, wsD, wsI);
  knn_final_kernel<<<KNN_B, 256, 0, stream>>>(X, labels, wsD, wsI, out);
}
